// GNSLayer_29592324670080
// MI455X (gfx1250) — matrix operand order repaired and checked by reading, not compiled, not run on hardware
//
#include <hip/hip_runtime.h>
#include <hip/hip_bf16.h>
#include <stdint.h>

#define N_NODES  50000
#define N_EDGES  500000
#define NODE_DIM 128
#define EDGE_DIM 64
#define HIDDEN   128
#define K_EDGE1  320            // 2*NODE_DIM + EDGE_DIM
#define K_NODE1  256            // NODE_DIM + HIDDEN
#define KS_E1    328            // padded LDS K-stride (bf16 elems) for eW1t
#define KS_N1    264            // padded LDS K-stride for nW1t
#define KS_128   136            // padded LDS K-stride for 128-deep weights / bf16 h-scratch
#define SCR_STRIDE 132          // f32 row stride for 16x128 wave scratch

typedef __bf16 v16bf __attribute__((ext_vector_type(16)));
typedef __bf16 v8bf  __attribute__((ext_vector_type(8)));
typedef float  v8f   __attribute__((ext_vector_type(8)));

union BFrag { v16bf v; v8bf b8[2]; uint32_t u[8]; };

// 8 f32 -> 8 bf16 via native vector fptrunc (v_cvt_pk_bf16_f32)
__device__ __forceinline__ v8bf cvt8(float4 a, float4 b) {
  v8f t;
  t[0]=a.x; t[1]=a.y; t[2]=a.z; t[3]=a.w;
  t[4]=b.x; t[5]=b.y; t[6]=b.z; t[7]=b.w;
  return __builtin_convertvector(t, v8bf);
}

__device__ __forceinline__ void load_chunk_f32(BFrag& f, int h, const float* p) {
  float4 a = ((const float4*)p)[0];
  float4 b = ((const float4*)p)[1];
  f.b8[h] = cvt8(a, b);
}

__device__ __forceinline__ void load_chunk_bf16(BFrag& f, int h, const __bf16* p) {
  f.b8[h] = *(const v8bf*)p;                     // single ds_load_b128
}

__device__ __forceinline__ void load_bfrag(BFrag& f, const __bf16* w, int idx) {
  const uint4* p = (const uint4*)(w + idx);
  uint4 a = p[0], b = p[2];
  f.u[0]=a.x; f.u[1]=a.y; f.u[2]=a.z; f.u[3]=a.w;
  f.u[4]=b.x; f.u[5]=b.y; f.u[6]=b.z; f.u[7]=b.w;
}

__device__ __forceinline__ v8f wmma_bf16(const BFrag& a, const BFrag& b, v8f c) {
  return __builtin_amdgcn_wmma_f32_16x16x32_bf16(false, a.v, false, b.v,
                                                 (short)0, c, false, false);
}

// ---------------------------------------------------------------------------
extern "C" __global__ void gns_zero_kernel(float4* __restrict__ p, int n4) {
  int i = blockIdx.x * blockDim.x + threadIdx.x;
  if (i < n4) p[i] = make_float4(0.f, 0.f, 0.f, 0.f);
}

// ---------------------------------------------------------------------------
// Edge kernel, MT=2: each wave handles 32 edges (two 16-row tiles); each
// B-fragment feeds two WMMAs and is double-buffered to hide LDS latency.
// ---------------------------------------------------------------------------
extern "C" __global__ __launch_bounds__(256)
void gns_edge_kernel(const float* __restrict__ x,
                     const long long* __restrict__ senders,
                     const long long* __restrict__ receivers,
                     const float* __restrict__ edge_attr,
                     const float* __restrict__ eW1, const float* __restrict__ eb1,
                     const float* __restrict__ eW2, const float* __restrict__ eb2,
                     const float* __restrict__ eg,  const float* __restrict__ ebeta,
                     float* __restrict__ agg)
{
  extern __shared__ char smem[];
  __bf16* w1 = (__bf16*)smem;                                          // [128][KS_E1]
  __bf16* w2 = (__bf16*)(smem + (size_t)128*KS_E1*2);                  // [128][KS_128]
  __bf16* hbase = (__bf16*)(smem + (size_t)128*KS_E1*2 + (size_t)128*KS_128*2);   // 8 x [32][KS_128]
  float*  fbase = (float*)(smem + (size_t)128*KS_E1*2 + (size_t)128*KS_128*2
                                + (size_t)8*32*KS_128*2);              // 8 x [16][SCR_STRIDE]

  const int tid = threadIdx.x;
  for (int i = tid; i < K_EDGE1*128; i += 256) {
    int k = i >> 7, n = i & 127;
    w1[n*KS_E1 + k] = (__bf16)eW1[i];
  }
  for (int i = tid; i < 128*128; i += 256) {
    int k = i >> 7, n = i & 127;
    w2[n*KS_128 + k] = (__bf16)eW2[i];
  }
  __syncthreads();

  const int wid  = tid >> 5;
  const int lane = tid & 31;
  const int tile = blockIdx.x * 8 + wid;
  if (tile >= N_EDGES/32) return;                 // 15625 exact; wave-uniform

  __bf16* hscr = hbase + wid * (32 * KS_128);     // wave-private 32x128 bf16 h scratch
  float*  fscr = fbase + wid * (16 * SCR_STRIDE); // wave-private 16x128 f32 scratch (reused per tile)
  const int m    = lane & 15;
  const int half = lane >> 4;
  const int koffA = half ? 8 : 0;
  const int koffB = half ? 8 : 0;
  const int hro  = half * 8;

  const int e0 = tile * 32 + m;
  const int e1 = e0 + 16;
  const long long s0 = senders[e0],   s1 = senders[e1];
  const long long r0 = receivers[e0], r1 = receivers[e1];
  const float* xs0 = x + (size_t)s0 * NODE_DIM;
  const float* xr0 = x + (size_t)r0 * NODE_DIM;
  const float* ea0 = edge_attr + (size_t)e0 * EDGE_DIM;
  const float* xs1 = x + (size_t)s1 * NODE_DIM;
  const float* xr1 = x + (size_t)r1 * NODE_DIM;
  const float* ea1 = edge_attr + (size_t)e1 * EDGE_DIM;

  // ---- Layer 1: 2 x [16 x 320] @ [320 x 128], B shared + double-buffered ----
  v8f acc0[8], acc1[8];
  #pragma unroll
  for (int nt = 0; nt < 8; nt++)
    #pragma unroll
    for (int j = 0; j < 8; j++) { acc0[nt][j] = 0.f; acc1[nt][j] = 0.f; }

  #pragma unroll
  for (int kc = 0; kc < 10; kc++) {
    const int kb = kc * 32;
    const float *p0, *p1; int off;
    if (kb < 128)      { p0 = xs0; p1 = xs1; off = kb; }
    else if (kb < 256) { p0 = xr0; p1 = xr1; off = kb - 128; }
    else               { p0 = ea0; p1 = ea1; off = kb - 256; }
    BFrag a0, a1;
    load_chunk_f32(a0, 0, p0 + off + koffA);
    load_chunk_f32(a0, 1, p0 + off + koffA + 16);
    load_chunk_f32(a1, 0, p1 + off + koffA);
    load_chunk_f32(a1, 1, p1 + off + koffA + 16);
    BFrag b[2];
    load_bfrag(b[0], w1, m * KS_E1 + kb + koffB);           // nt = 0
    #pragma unroll
    for (int nt = 0; nt < 8; nt++) {
      if (nt < 7)
        load_bfrag(b[(nt+1)&1], w1, ((nt+1)*16 + m) * KS_E1 + kb + koffB);
      acc0[nt] = wmma_bf16(a0, b[nt&1], acc0[nt]);
      acc1[nt] = wmma_bf16(a1, b[nt&1], acc1[nt]);
    }
  }
  // bias + relu -> bf16 scratch (rows 0-15 = tile0, rows 16-31 = tile1)
  #pragma unroll
  for (int nt = 0; nt < 8; nt++) {
    float bias = eb1[nt*16 + m];
    #pragma unroll
    for (int j = 0; j < 8; j++) {
      hscr[(hro + j)      * KS_128 + nt*16 + m] = (__bf16)fmaxf(acc0[nt][j] + bias, 0.f);
      hscr[(16 + hro + j) * KS_128 + nt*16 + m] = (__bf16)fmaxf(acc1[nt][j] + bias, 0.f);
    }
  }

  // ---- Layer 2: 2 x [16 x 128] @ [128 x 128], B shared + double-buffered ----
  v8f acc20[8], acc21[8];
  #pragma unroll
  for (int nt = 0; nt < 8; nt++)
    #pragma unroll
    for (int j = 0; j < 8; j++) { acc20[nt][j] = 0.f; acc21[nt][j] = 0.f; }

  #pragma unroll
  for (int kc = 0; kc < 4; kc++) {
    BFrag a0, a1;
    const __bf16* q0 = hscr + m * KS_128 + kc*32 + koffA;
    const __bf16* q1 = hscr + (16 + m) * KS_128 + kc*32 + koffA;
    load_chunk_bf16(a0, 0, q0);
    load_chunk_bf16(a0, 1, q0 + 16);
    load_chunk_bf16(a1, 0, q1);
    load_chunk_bf16(a1, 1, q1 + 16);
    BFrag b[2];
    load_bfrag(b[0], w2, m * KS_128 + kc*32 + koffB);
    #pragma unroll
    for (int nt = 0; nt < 8; nt++) {
      if (nt < 7)
        load_bfrag(b[(nt+1)&1], w2, ((nt+1)*16 + m) * KS_128 + kc*32 + koffB);
      acc20[nt] = wmma_bf16(a0, b[nt&1], acc20[nt]);
      acc21[nt] = wmma_bf16(a1, b[nt&1], acc21[nt]);
    }
  }

  // ---- per-tile: bias + relu -> f32 scratch, LayerNorm, atomic scatter ----
  #pragma unroll
  for (int t = 0; t < 2; t++) {
    v8f* acc2 = t ? acc21 : acc20;
    const long long recv = t ? r1 : r0;
    #pragma unroll
    for (int nt = 0; nt < 8; nt++) {
      float bias = eb2[nt*16 + m];
      #pragma unroll
      for (int j = 0; j < 8; j++)
        fscr[(hro + j) * SCR_STRIDE + nt*16 + m] = fmaxf(acc2[nt][j] + bias, 0.f);
    }
    const float* row = fscr + m * SCR_STRIDE + half * 64;
    float sum = 0.f, sq = 0.f;
    #pragma unroll
    for (int j = 0; j < 16; j++) {
      float4 v = ((const float4*)row)[j];
      sum += v.x + v.y + v.z + v.w;
      sq  += v.x*v.x + v.y*v.y + v.z*v.z + v.w*v.w;
    }
    sum += __shfl_xor(sum, 16);
    sq  += __shfl_xor(sq, 16);
    const float mean = sum * (1.f / 128.f);
    const float var  = sq * (1.f / 128.f) - mean * mean;
    const float rs   = rsqrtf(var + 1e-5f);

    float* aggrow = agg + (size_t)recv * HIDDEN + half * 64;
    const float4* g4 = (const float4*)eg    + half * 16;
    const float4* b4 = (const float4*)ebeta + half * 16;
    #pragma unroll
    for (int j = 0; j < 16; j++) {
      float4 v = ((const float4*)row)[j];
      float4 g = g4[j], bb = b4[j];
      unsafeAtomicAdd(aggrow + j*4 + 0, (v.x - mean) * rs * g.x + bb.x);
      unsafeAtomicAdd(aggrow + j*4 + 1, (v.y - mean) * rs * g.y + bb.y);
      unsafeAtomicAdd(aggrow + j*4 + 2, (v.z - mean) * rs * g.z + bb.z);
      unsafeAtomicAdd(aggrow + j*4 + 3, (v.w - mean) * rs * g.w + bb.w);
    }
  }
}

// ---------------------------------------------------------------------------
// Node kernel (MT=1): out = LN(x + (relu([x|agg] @ nW1 + nb1) @ nW2 + nb2))
// ---------------------------------------------------------------------------
extern "C" __global__ __launch_bounds__(256)
void gns_node_kernel(const float* __restrict__ x, const float* __restrict__ agg,
                     const float* __restrict__ nW1, const float* __restrict__ nb1,
                     const float* __restrict__ nW2, const float* __restrict__ nb2,
                     const float* __restrict__ ng,  const float* __restrict__ nbeta,
                     float* __restrict__ out)
{
  extern __shared__ char smem[];
  __bf16* w1 = (__bf16*)smem;                                          // [128][KS_N1]
  __bf16* w2 = (__bf16*)(smem + (size_t)128*KS_N1*2);                  // [128][KS_128]
  __bf16* hbase = (__bf16*)(smem + (size_t)128*KS_N1*2 + (size_t)128*KS_128*2);
  float*  fbase = (float*)(smem + (size_t)128*KS_N1*2 + (size_t)128*KS_128*2
                                + (size_t)8*16*KS_128*2);

  const int tid = threadIdx.x;
  for (int i = tid; i < K_NODE1*128; i += 256) {
    int k = i >> 7, n = i & 127;
    w1[n*KS_N1 + k] = (__bf16)nW1[i];
  }
  for (int i = tid; i < 128*128; i += 256) {
    int k = i >> 7, n = i & 127;
    w2[n*KS_128 + k] = (__bf16)nW2[i];
  }
  __syncthreads();

  const int wid  = tid >> 5;
  const int lane = tid & 31;
  const int tile = blockIdx.x * 8 + wid;
  if (tile >= N_NODES/16) return;

  __bf16* hscr = hbase + wid * (16 * KS_128);
  float*  fscr = fbase + wid * (16 * SCR_STRIDE);
  const int m = lane & 15, half = lane >> 4;
  const int koffA = half ? 8 : 0, koffB = half ? 8 : 0, hro = half * 8;

  const int node = tile * 16 + m;
  const float* xn = x   + (size_t)node * NODE_DIM;
  const float* an = agg + (size_t)node * HIDDEN;

  v8f acc[8];
  #pragma unroll
  for (int nt = 0; nt < 8; nt++)
    #pragma unroll
    for (int j = 0; j < 8; j++) acc[nt][j] = 0.f;

  #pragma unroll
  for (int kc = 0; kc < 8; kc++) {
    const int kb = kc * 32;
    const float* src = (kb < 128) ? (xn + kb) : (an + kb - 128);
    BFrag a;
    load_chunk_f32(a, 0, src + koffA);
    load_chunk_f32(a, 1, src + koffA + 16);
    BFrag b[2];
    load_bfrag(b[0], w1, m * KS_N1 + kb + koffB);
    #pragma unroll
    for (int nt = 0; nt < 8; nt++) {
      if (nt < 7)
        load_bfrag(b[(nt+1)&1], w1, ((nt+1)*16 + m) * KS_N1 + kb + koffB);
      acc[nt] = wmma_bf16(a, b[nt&1], acc[nt]);
    }
  }
  #pragma unroll
  for (int nt = 0; nt < 8; nt++) {
    float bias = nb1[nt*16 + m];
    #pragma unroll
    for (int j = 0; j < 8; j++)
      hscr[(hro + j) * KS_128 + nt*16 + m] = (__bf16)fmaxf(acc[nt][j] + bias, 0.f);
  }

  v8f acc2[8];
  #pragma unroll
  for (int nt = 0; nt < 8; nt++)
    #pragma unroll
    for (int j = 0; j < 8; j++) acc2[nt][j] = 0.f;

  #pragma unroll
  for (int kc = 0; kc < 4; kc++) {
    BFrag a;
    const __bf16* p = hscr + m * KS_128 + kc*32 + koffA;
    load_chunk_bf16(a, 0, p);
    load_chunk_bf16(a, 1, p + 16);
    BFrag b[2];
    load_bfrag(b[0], w2, m * KS_128 + kc*32 + koffB);
    #pragma unroll
    for (int nt = 0; nt < 8; nt++) {
      if (nt < 7)
        load_bfrag(b[(nt+1)&1], w2, ((nt+1)*16 + m) * KS_128 + kc*32 + koffB);
      acc2[nt] = wmma_bf16(a, b[nt&1], acc2[nt]);
    }
  }
  #pragma unroll
  for (int nt = 0; nt < 8; nt++) {
    float bias = nb2[nt*16 + m];
    #pragma unroll
    for (int j = 0; j < 8; j++)
      fscr[(hro + j) * SCR_STRIDE + nt*16 + m] = acc2[nt][j] + bias;
  }

  float* rowp = fscr + m * SCR_STRIDE + half * 64;
  const float4* xrow = (const float4*)(xn) + half * 16;
  float sum = 0.f, sq = 0.f;
  #pragma unroll
  for (int j = 0; j < 16; j++) {
    float4 u = ((float4*)rowp)[j];
    float4 xv = xrow[j];
    float4 t = make_float4(u.x + xv.x, u.y + xv.y, u.z + xv.z, u.w + xv.w);
    ((float4*)rowp)[j] = t;
    sum += t.x + t.y + t.z + t.w;
    sq  += t.x*t.x + t.y*t.y + t.z*t.z + t.w*t.w;
  }
  sum += __shfl_xor(sum, 16);
  sq  += __shfl_xor(sq, 16);
  const float mean = sum * (1.f / 128.f);
  const float var  = sq * (1.f / 128.f) - mean * mean;
  const float rs   = rsqrtf(var + 1e-5f);

  float4* orow = (float4*)(out + (size_t)node * NODE_DIM) + half * 16;
  const float4* g4 = (const float4*)ng    + half * 16;
  const float4* b4 = (const float4*)nbeta + half * 16;
  #pragma unroll
  for (int j = 0; j < 16; j++) {
    float4 t = ((float4*)rowp)[j];
    float4 g = g4[j], bb = b4[j];
    orow[j] = make_float4((t.x - mean) * rs * g.x + bb.x,
                          (t.y - mean) * rs * g.y + bb.y,
                          (t.z - mean) * rs * g.z + bb.z,
                          (t.w - mean) * rs * g.w + bb.w);
  }
}

// ---------------------------------------------------------------------------
extern "C" void kernel_launch(void* const* d_in, const int* in_sizes, int n_in,
                              void* d_out, int out_size, void* d_ws, size_t ws_size,
                              hipStream_t stream) {
  (void)in_sizes; (void)n_in; (void)out_size; (void)ws_size;
  const float*     x         = (const float*)d_in[0];
  const long long* eidx      = (const long long*)d_in[1];   // int64 [2, E]
  const float*     edge_attr = (const float*)d_in[2];
  const float*     eW1 = (const float*)d_in[3];
  const float*     eb1 = (const float*)d_in[4];
  const float*     eW2 = (const float*)d_in[5];
  const float*     eb2 = (const float*)d_in[6];
  const float*     nW1 = (const float*)d_in[7];
  const float*     nb1 = (const float*)d_in[8];
  const float*     nW2 = (const float*)d_in[9];
  const float*     nb2 = (const float*)d_in[10];
  const float*     eg    = (const float*)d_in[11];
  const float*     ebeta = (const float*)d_in[12];
  const float*     ng    = (const float*)d_in[13];
  const float*     nbeta = (const float*)d_in[14];

  const long long* senders   = eidx;
  const long long* receivers = eidx + N_EDGES;
  float* agg = (float*)d_ws;                     // [N_NODES][HIDDEN] f32
  float* out = (float*)d_out;

  const size_t smem_e = (size_t)128*KS_E1*2 + (size_t)128*KS_128*2
                      + (size_t)8*32*KS_128*2 + (size_t)8*16*SCR_STRIDE*4;   // 256000 B
  const size_t smem_n = (size_t)128*KS_N1*2 + (size_t)128*KS_128*2
                      + (size_t)8*16*KS_128*2 + (size_t)8*16*SCR_STRIDE*4;
  hipFuncSetAttribute(reinterpret_cast<const void*>(gns_edge_kernel),
                      hipFuncAttributeMaxDynamicSharedMemorySize, (int)smem_e);
  hipFuncSetAttribute(reinterpret_cast<const void*>(gns_node_kernel),
                      hipFuncAttributeMaxDynamicSharedMemorySize, (int)smem_n);

  const int n4 = N_NODES * HIDDEN / 4;
  gns_zero_kernel<<<(n4 + 255) / 256, 256, 0, stream>>>((float4*)agg, n4);

  const int etiles = N_EDGES / 32;               // 15625 exact (32 edges per wave)
  gns_edge_kernel<<<(etiles + 7) / 8, 256, smem_e, stream>>>(
      x, senders, receivers, edge_attr, eW1, eb1, eW2, eb2, eg, ebeta, agg);

  const int ntiles = N_NODES / 16;               // 3125 exact
  gns_node_kernel<<<(ntiles + 7) / 8, 256, smem_n, stream>>>(
      x, agg, nW1, nb1, nW2, nb2, ng, nbeta, out);
}
